// NMILoss_84284438217027
// MI455X (gfx1250) — matrix operand order repaired and checked by reading, not compiled, not run on hardware
//
#include <hip/hip_runtime.h>
#include <math.h>

// ---------------- problem constants ----------------
#define NUM_BINS 64
#define N_VOX    (64 * 64 * 64)   // 262144 voxels per batch
#define BATCH    4
#define TINY     1e-12f
// hist = einsum / EPS^2 ; EPS = 1/64 -> scale = 4096
#define HIST_SCALE 4096.0f

// ---------------- launch-shape constants ----------------
#define BPB      64    // histogram blocks per batch
#define WAVES    4     // waves per hist block (128 threads)
#define CHUNK    32    // K samples per WMMA step
#define MMBLOCKS 256   // min/max partial blocks

typedef __attribute__((ext_vector_type(16))) _Float16 v16h;
typedef __attribute__((ext_vector_type(8)))  float    v8f;

// Cubic B-spline kernel, branchless.
__device__ __forceinline__ float bspline3(float t) {
  float a  = __builtin_fabsf(t);
  float a2 = a * a;
  float inner = 0.66666667f - a2 + 0.5f * a2 * a;   // |t| < 1
  float om = 2.0f - a;
  float outer = om * om * om * (1.0f / 6.0f);       // 1 <= |t| < 2
  float w = (a < 1.0f) ? inner : outer;
  return (a < 2.0f) ? w : 0.0f;
}

// =====================================================================
// Kernel 1: per-block min/max partials over BOTH tensors (global norm)
// =====================================================================
__global__ void nmi_minmax_partial(const float* __restrict__ t,
                                   const float* __restrict__ s,
                                   int n, float* __restrict__ part) {
  __shared__ float r0[256], r1[256], r2[256], r3[256];
  int tid = threadIdx.x;
  float mnt = 3.4e38f, mxt = -3.4e38f, mns = 3.4e38f, mxs = -3.4e38f;
  for (int i = blockIdx.x * blockDim.x + tid; i < n; i += gridDim.x * blockDim.x) {
    float a = t[i]; mnt = fminf(mnt, a); mxt = fmaxf(mxt, a);
    float b = s[i]; mns = fminf(mns, b); mxs = fmaxf(mxs, b);
  }
  r0[tid] = mnt; r1[tid] = mxt; r2[tid] = mns; r3[tid] = mxs;
  __syncthreads();
  for (int off = 128; off > 0; off >>= 1) {
    if (tid < off) {
      r0[tid] = fminf(r0[tid], r0[tid + off]);
      r1[tid] = fmaxf(r1[tid], r1[tid + off]);
      r2[tid] = fminf(r2[tid], r2[tid + off]);
      r3[tid] = fmaxf(r3[tid], r3[tid + off]);
    }
    __syncthreads();
  }
  if (tid == 0) {
    part[blockIdx.x * 4 + 0] = r0[0];
    part[blockIdx.x * 4 + 1] = r1[0];
    part[blockIdx.x * 4 + 2] = r2[0];
    part[blockIdx.x * 4 + 3] = r3[0];
  }
}

// =====================================================================
// Kernel 2: final min/max reduction -> stats = {min_t, inv_t, min_s, inv_s}
// =====================================================================
__global__ void nmi_minmax_final(const float* __restrict__ part,
                                 float* __restrict__ stats) {
  __shared__ float r0[256], r1[256], r2[256], r3[256];
  int tid = threadIdx.x;   // MMBLOCKS == 256 == blockDim.x
  r0[tid] = part[tid * 4 + 0];
  r1[tid] = part[tid * 4 + 1];
  r2[tid] = part[tid * 4 + 2];
  r3[tid] = part[tid * 4 + 3];
  __syncthreads();
  for (int off = 128; off > 0; off >>= 1) {
    if (tid < off) {
      r0[tid] = fminf(r0[tid], r0[tid + off]);
      r1[tid] = fmaxf(r1[tid], r1[tid + off]);
      r2[tid] = fminf(r2[tid], r2[tid + off]);
      r3[tid] = fmaxf(r3[tid], r3[tid + off]);
    }
    __syncthreads();
  }
  if (tid == 0) {
    stats[0] = r0[0];
    stats[1] = 1.0f / (r1[0] - r0[0] + TINY);
    stats[2] = r2[0];
    stats[3] = 1.0f / (r3[0] - r2[0] + TINY);
  }
}

// =====================================================================
// Kernel 3: joint histogram as dense GEMM via V_WMMA_F32_16X16X32_F16.
//   A = W_t tile (16 bins x 32 samples), B = W_s^T tile (32 samples x 16 bins)
//
// Register-pressure layout: each wave owns HALF of the 64x64 output
// (2 row-blocks x 4 col-blocks = 8 accumulators = 64 VGPRs). The 4 waves
// of a block form 2 row-halves x 2 chunk-sets:
//   rp = wave & 1  -> rows rp*32 .. rp*32+31
//   cs = wave >> 1 -> chunks c == cs (mod 2)
// grid = (BPB, BATCH), block = 32*WAVES, dynSmem = 2*4096*4 bytes.
// =====================================================================
__global__ void __launch_bounds__(32 * WAVES, 1)
nmi_hist_wmma(const float* __restrict__ tgt,
              const float* __restrict__ src,
              const float* __restrict__ stats,
              float* __restrict__ partial) {
  extern __shared__ float smem[];              // 2 * 4096 floats (one hist per chunk-set)
  const int tid  = threadIdx.x;
  const int wave = tid >> 5;
  const int lane = tid & 31;
  const int r    = lane & 15;                  // A-row / B-col within 16-block
  const int hi   = lane >> 4;                  // half-wave selector
  const int rp   = wave & 1;                   // row half (bins rp*32 ..)
  const int cs   = wave >> 1;                  // chunk set
  const int b    = blockIdx.y;
  const int blk  = blockIdx.x;

  const float mn_t = stats[0], sc_t = stats[1] * (float)NUM_BINS;
  const float mn_s = stats[2], sc_s = stats[3] * (float)NUM_BINS;

  const float* t_base = tgt + (size_t)b * N_VOX;
  const float* s_base = src + (size_t)b * N_VOX;

  v8f acc[8];
#pragma unroll
  for (int i = 0; i < 8; ++i) acc[i] = (v8f){0, 0, 0, 0, 0, 0, 0, 0};

  const int samplesPerBlock = N_VOX / BPB;               // 4096
  const int chunksPerBlock  = samplesPerBlock / CHUNK;   // 128
  const int base0 = blk * samplesPerBlock;

  for (int c = cs; c < chunksPerBlock; c += 2) {
    const int s0 = base0 + c * CHUNK;

    // ---- source samples in B-fragment K order for this lane ----
    // lanes 0-15: K slots {0..7, 16..23}; lanes 16-31: {8..15, 24..31}
    float us[16];
    {
      const float4* p = (const float4*)(s_base + s0);
      float4 q;
      q = p[hi * 2 + 0]; us[0]  = q.x; us[1]  = q.y; us[2]  = q.z; us[3]  = q.w;
      q = p[hi * 2 + 1]; us[4]  = q.x; us[5]  = q.y; us[6]  = q.z; us[7]  = q.w;
      q = p[hi * 2 + 4]; us[8]  = q.x; us[9]  = q.y; us[10] = q.z; us[11] = q.w;
      q = p[hi * 2 + 5]; us[12] = q.x; us[13] = q.y; us[14] = q.z; us[15] = q.w;
#pragma unroll
      for (int j = 0; j < 16; ++j) us[j] = (us[j] - mn_s) * sc_s;  // bin units
    }

    // ---- B fragments: W_s^T, all 4 column blocks ----
    v16h bf[4];
#pragma unroll
    for (int bj = 0; bj < 4; ++bj) {
      float binc = (float)(bj * 16 + r);
#pragma unroll
      for (int j = 0; j < 16; ++j)
        bf[bj][j] = (_Float16)bspline3(binc - us[j]);
    }

    // ---- target samples in A-fragment K order for this lane ----
    // lanes 0-15: K slots {0..7, 16..23}; lanes 16-31: {8..15, 24..31}
    float ut[16];
    {
      const float4* p = (const float4*)(t_base + s0);
      float4 q;
      q = p[hi * 2 + 0]; ut[0]  = q.x; ut[1]  = q.y; ut[2]  = q.z; ut[3]  = q.w;
      q = p[hi * 2 + 1]; ut[4]  = q.x; ut[5]  = q.y; ut[6]  = q.z; ut[7]  = q.w;
      q = p[hi * 2 + 4]; ut[8]  = q.x; ut[9]  = q.y; ut[10] = q.z; ut[11] = q.w;
      q = p[hi * 2 + 5]; ut[12] = q.x; ut[13] = q.y; ut[14] = q.z; ut[15] = q.w;
#pragma unroll
      for (int j = 0; j < 16; ++j) ut[j] = (ut[j] - mn_t) * sc_t;
    }

    // ---- A fragments (this wave's 2 row blocks) + 8 WMMAs ----
#pragma unroll
    for (int biL = 0; biL < 2; ++biL) {
      float binr = (float)((rp * 2 + biL) * 16 + r);
      v16h af;
#pragma unroll
      for (int j = 0; j < 16; ++j)
        af[j] = (_Float16)bspline3(binr - ut[j]);
#pragma unroll
      for (int bj = 0; bj < 4; ++bj) {
        acc[biL * 4 + bj] = __builtin_amdgcn_wmma_f32_16x16x32_f16(
            false, af, false, bf[bj], (short)0, acc[biL * 4 + bj], false, false);
      }
    }
  }

  // ---- spill each wave's 32x64 half-hist to its LDS region ----
  // C/D layout: VGPR v -> row M = v + 8*hi, lane -> col N = r.
  // The two waves sharing a chunk-set write disjoint row halves: plain stores.
  float* wh = smem + cs * (NUM_BINS * NUM_BINS);
#pragma unroll
  for (int biL = 0; biL < 2; ++biL)
#pragma unroll
    for (int bj = 0; bj < 4; ++bj)
#pragma unroll
      for (int vg = 0; vg < 8; ++vg) {
        int row = (rp * 2 + biL) * 16 + vg + 8 * hi;
        int col = bj * 16 + r;
        wh[row * NUM_BINS + col] = acc[biL * 4 + bj][vg];
      }
  __syncthreads();

  // ---- deterministic fixed-order sum across the 2 chunk-sets ----
  float* out = partial + (size_t)(b * BPB + blk) * (NUM_BINS * NUM_BINS);
  for (int e = tid; e < NUM_BINS * NUM_BINS; e += blockDim.x)
    out[e] = smem[e] + smem[NUM_BINS * NUM_BINS + e];
}

// =====================================================================
// Kernel 4: fixed-order global reduction + entropies -> scalar loss.
// Single block of 256 threads.
// =====================================================================
__device__ __forceinline__ float block_reduce_sum256(float v, float* red) {
  int tid = threadIdx.x;
  red[tid] = v;
  __syncthreads();
  for (int off = 128; off > 0; off >>= 1) {
    if (tid < off) red[tid] += red[tid + off];
    __syncthreads();
  }
  float r = red[0];
  __syncthreads();
  return r;
}

__global__ void nmi_finalize(const float* __restrict__ partial,
                             float* __restrict__ sumhist,
                             float* __restrict__ out) {
  __shared__ float red[256];
  const int tid = threadIdx.x;
  const int HB = NUM_BINS * NUM_BINS;   // 4096

  // Phase A: sum partials per batch (fixed order), scale, track grand total.
  float lt = 0.0f;
  for (int b = 0; b < BATCH; ++b) {
    const float* p = partial + (size_t)b * BPB * HB;
    for (int e = tid; e < HB; e += 256) {
      float s = 0.0f;
      for (int k = 0; k < BPB; ++k) s += p[(size_t)k * HB + e];
      s *= HIST_SCALE;
      sumhist[b * HB + e] = s;
      lt += s;
    }
  }
  float total = block_reduce_sum256(lt, red);   // also fences sumhist writes
  float inv_total = 1.0f / total;

  // Phase B: entropies per batch.
  float ratio_sum = 0.0f;
  for (int b = 0; b < BATCH; ++b) {
    const float* h = sumhist + b * HB;

    float hj_l = 0.0f;
    for (int e = tid; e < HB; e += 256) {
      float pj = h[e] * inv_total;
      hj_l += pj * logf(pj + TINY);
    }
    float ht_l = 0.0f, hs_l = 0.0f;
    if (tid < NUM_BINS) {
      float pt = 0.0f, ps = 0.0f;
      for (int j = 0; j < NUM_BINS; ++j) {
        pt += h[tid * NUM_BINS + j];   // row sum   -> p_target
        ps += h[j * NUM_BINS + tid];   // col sum   -> p_source
      }
      pt *= inv_total; ps *= inv_total;
      ht_l = pt * logf(pt + TINY);
      hs_l = ps * logf(ps + TINY);
    }
    float Hj = -block_reduce_sum256(hj_l, red);
    float Ht = -block_reduce_sum256(ht_l, red);
    float Hs = -block_reduce_sum256(hs_l, red);
    ratio_sum += (Ht + Hs) / Hj;
  }
  if (tid == 0) out[0] = -(ratio_sum * (1.0f / (float)BATCH));
}

// =====================================================================
// Host entry
// =====================================================================
extern "C" void kernel_launch(void* const* d_in, const int* in_sizes, int n_in,
                              void* d_out, int out_size, void* d_ws, size_t ws_size,
                              hipStream_t stream) {
  const float* tgt = (const float*)d_in[0];
  const float* src = (const float*)d_in[1];
  float* out = (float*)d_out;
  float* ws  = (float*)d_ws;

  // Workspace layout (floats):
  float* stats   = ws;             //     4 : {min_t, inv_t, min_s, inv_s}
  float* mmpart  = ws + 8;         //  1024 : MMBLOCKS*4 min/max partials
  float* sumhist = ws + 2048;      // 16384 : BATCH*4096 summed histograms
  float* partial = ws + 32768;     // BATCH*BPB*4096 block partials (~4 MB)

  const int ntot = BATCH * N_VOX;

  nmi_minmax_partial<<<MMBLOCKS, 256, 0, stream>>>(tgt, src, ntot, mmpart);
  nmi_minmax_final<<<1, 256, 0, stream>>>(mmpart, stats);

  dim3 grid(BPB, BATCH);
  size_t smem = (size_t)2 * NUM_BINS * NUM_BINS * sizeof(float);  // 32 KB
  nmi_hist_wmma<<<grid, 32 * WAVES, smem, stream>>>(tgt, src, stats, partial);

  nmi_finalize<<<1, 256, 0, stream>>>(partial, sumhist, out);
}
